// MultiScaleQuantizer2d_65463891526267
// MI455X (gfx1250) — matrix operand order repaired and checked by reading, not compiled, not run on hardware
//
#include <hip/hip_runtime.h>

// ---------------------------------------------------------------------------
// MultiScaleQuantizer2d for MI455X (gfx1250, wave32, WMMA).
// Heavy matmul (flat @ cb_n^T, up to 4096x256x8192 per scale) runs on
// v_wmma_f32_16x16x32_bf16 with codebook tiles staged into LDS via
// global_load_async_to_lds_b128 (double-buffered, shared by 4 waves per WGP
// workgroup -> 4x less L2 traffic). All statistics / EMA math stays fp32 and
// is fully deterministic (no float atomics).
// Workspace requirement: ~41 MB (10,117,601 floats).
// ---------------------------------------------------------------------------

typedef __attribute__((ext_vector_type(16))) __bf16        v16bf;
typedef __attribute__((ext_vector_type(8)))  float         v8f;
typedef __attribute__((ext_vector_type(4)))  unsigned int  u32x4;

union FragBF16 { u32x4 q[2]; v16bf v; };

static __device__ __forceinline__ unsigned short f2bf(float f) {
  unsigned u = __float_as_uint(f);
  unsigned r = u + 0x7FFFu + ((u >> 16) & 1u);   // round-to-nearest-even
  return (unsigned short)(r >> 16);
}

// ---------------- constants ----------------
#define KC   8192      // vocab
#define DC   256       // codebook dim
#define HWC  64        // H == W
#define NCHUNK 16      // N-chunks in score kernel (8192/16 = 512 cols/chunk)
#define NTILES_PER_CHUNK 32   // 512 / 16

// ---------------------------------------------------------------------------
// 1) Bilinear (antialiased triangle, matches jax.image.resize) downsample
// ---------------------------------------------------------------------------
__global__ void resize_kernel(const float* __restrict__ src, float* __restrict__ dst,
                              int sh, int sw) {
  int S = sh * sw;
  int N = DC * S;
  int t = blockIdx.x * blockDim.x + threadIdx.x;
  if (t >= N) return;
  int d = t / S, p = t % S;
  int i = p / sw, j = p % sw;
  float scale = (float)sh * (1.0f / 64.0f);   // <= 1 (downsample)
  float rad   = 64.0f / (float)sh;            // kernel radius 1/scale
  float cy = ((float)i + 0.5f) / scale - 0.5f;
  float cx = ((float)j + 0.5f) / scale - 0.5f;
  int y0 = (int)ceilf(cy - rad);  if (y0 < 0)  y0 = 0;
  int y1 = (int)floorf(cy + rad); if (y1 > 63) y1 = 63;
  int x0 = (int)ceilf(cx - rad);  if (x0 < 0)  x0 = 0;
  int x1 = (int)floorf(cx + rad); if (x1 > 63) x1 = 63;
  float wx_sum = 0.f;
  for (int x = x0; x <= x1; ++x) {
    float wx = 1.0f - fabsf(((float)x - cx) * scale);
    if (wx > 0.f) wx_sum += wx;
  }
  float wy_sum = 0.f, num = 0.f;
  const float* sd = src + (size_t)d * 4096;
  for (int y = y0; y <= y1; ++y) {
    float wy = 1.0f - fabsf(((float)y - cy) * scale);
    if (wy <= 0.f) continue;
    wy_sum += wy;
    float row = 0.f;
    for (int x = x0; x <= x1; ++x) {
      float wx = 1.0f - fabsf(((float)x - cx) * scale);
      if (wx > 0.f) row += wx * sd[y * 64 + x];
    }
    num += wy * row;
  }
  dst[t] = num / (wy_sum * wx_sum);
}

// ---------------------------------------------------------------------------
// 2) 3x3 conv, pad=1, OIHW weights.
// ---------------------------------------------------------------------------
__global__ void conv3x3_kernel(const float* __restrict__ in, const float* __restrict__ w,
                               const float* __restrict__ bias, float* __restrict__ outp,
                               int sh, int sw) {
  __shared__ float patch[9 * DC];
  int p = blockIdx.x;                 // pixel
  int c = threadIdx.x;                // channel (stage) / out-channel (compute)
  int y = p / sw, x = p % sw;
  int S = sh * sw;
  #pragma unroll
  for (int t = 0; t < 9; ++t) {
    int yy = y + t / 3 - 1, xx = x + t % 3 - 1;
    float v = 0.f;
    if (yy >= 0 && yy < sh && xx >= 0 && xx < sw) v = in[(size_t)c * S + yy * sw + xx];
    patch[t * DC + c] = v;
  }
  __syncthreads();
  const float* wrow = w + (size_t)c * (DC * 9);
  float acc = bias[c];
  for (int ci = 0; ci < DC; ++ci) {
    const float* wc = wrow + ci * 9;
    #pragma unroll
    for (int t = 0; t < 9; ++t) acc += patch[t * DC + ci] * wc[t];
  }
  outp[(size_t)c * S + p] = acc;
}

// ---------------------------------------------------------------------------
// 3) L2-normalize rows of flat, fp32 + bf16 copies; pad rows zeroed.
// ---------------------------------------------------------------------------
__global__ void norm_flat_kernel(const float* __restrict__ rk, float* __restrict__ flat,
                                 unsigned short* __restrict__ flatb, int P, int S) {
  int p = blockIdx.x, d = threadIdx.x;
  float v = (p < P) ? rk[(size_t)d * S + p] : 0.f;
  __shared__ float red[DC];
  red[d] = v * v; __syncthreads();
  for (int t = 128; t > 0; t >>= 1) { if (d < t) red[d] += red[d + t]; __syncthreads(); }
  float o = v / (sqrtf(red[0]) + 1e-8f);
  if (p >= P) o = 0.f;
  flat[(size_t)p * DC + d]  = o;
  flatb[(size_t)p * DC + d] = f2bf(o);
}

// 4) normalize codebook rows (fp32 + bf16)
__global__ void cbnorm_kernel(const float* __restrict__ cb, float* __restrict__ cbn,
                              unsigned short* __restrict__ cbnb) {
  int n = blockIdx.x, d = threadIdx.x;
  float v = cb[(size_t)n * DC + d];
  __shared__ float red[DC];
  red[d] = v * v; __syncthreads();
  for (int t = 128; t > 0; t >>= 1) { if (d < t) red[d] += red[d + t]; __syncthreads(); }
  float o = v / (sqrtf(red[0]) + 1e-8f);
  cbn[(size_t)n * DC + d]  = o;
  cbnb[(size_t)n * DC + d] = f2bf(o);
}

// ---------------------------------------------------------------------------
// 5) WMMA scoring + per-chunk argmax.
//    Workgroup = 128 threads (4 waves). Each wave owns one M-tile (16 rows);
//    the 16-column B tile (16 x 256 bf16 = 8 KB) is async-loaded into LDS
//    (double-buffered) and shared by all 4 waves.
//    A frag (16x32 bf16): lane m=lane&15 holds row m; K-halves by lane>>4.
//    B frag (32x16 bf16): lane holds column n=lane&15, K-half by lane>>4.
//    D (16x16 f32): acc[r] = score[row r + 8*(lane>=16)][col = lane&15].
// ---------------------------------------------------------------------------
static __device__ __forceinline__ void issue_tile_async(
    const unsigned short* __restrict__ cbb, int n0, unsigned ldsBase, int tid) {
  // tile = 16 rows x 512 bytes = 512 x 16B chunks; 128 threads x 4 chunks
  #pragma unroll
  for (int it = 0; it < 4; ++it) {
    int chunk = tid + it * 128;
    int row = chunk >> 5;                   // 0..15 (codebook row n0+row)
    int off = (chunk & 31) << 4;            // byte offset within the 512B row
    unsigned long long g =
        (unsigned long long)(size_t)(cbb + (size_t)(n0 + row) * DC) + (unsigned)off;
    unsigned lds_off = ldsBase + ((unsigned)chunk << 4);
    asm volatile("global_load_async_to_lds_b128 %0, %1, off"
                 :: "v"(lds_off), "v"(g) : "memory");
  }
}

__global__ __launch_bounds__(128)
void score_argmax_kernel(const unsigned short* __restrict__ flatb,
                         const unsigned short* __restrict__ cbb,
                         float* __restrict__ pval, int* __restrict__ pidx, int P) {
  __shared__ __align__(16) unsigned short ldsB[2][16 * DC];   // 2 x 8 KB

  const int tid   = threadIdx.x;        // 0..127
  const int lane  = tid & 31;
  const int wid   = tid >> 5;           // wave 0..3
  const int m     = lane & 15;
  const int half  = lane >> 4;
  const int chunk = blockIdx.x;         // 0..NCHUNK-1
  const int mtile = blockIdx.y * 4 + wid;
  const int n0base = chunk * (KC / NCHUNK);

  const unsigned ldsBase = (unsigned)(size_t)&ldsB[0][0];   // flat->LDS: low 32 bits

  // preload all 8 A k-step fragments (D=256, K-step 32); phantom mtiles read
  // stale-but-allocated rows of the 4096-row flatb buffer (results discarded).
  FragBF16 a[8];
  {
    const unsigned short* arow = flatb + ((size_t)(mtile * 16 + m)) * DC;
    #pragma unroll
    for (int ks = 0; ks < 8; ++ks) {
      const unsigned short* p0 = arow + ks * 32 + half * 8;
      a[ks].q[0] = *reinterpret_cast<const u32x4*>(p0);
      a[ks].q[1] = *reinterpret_cast<const u32x4*>(p0 + 16);
    }
  }

  float bestv[8]; int besti[8];
  #pragma unroll
  for (int r = 0; r < 8; ++r) { bestv[r] = -3.0e38f; besti[r] = 0; }

  // prologue: async-load tile 0 into buffer 0
  issue_tile_async(cbb, n0base, ldsBase, tid);

  for (int nt = 0; nt < NTILES_PER_CHUNK; ++nt) {
    // tile nt resident + all waves done reading tile nt-1
    asm volatile("s_wait_asynccnt 0x0" ::: "memory");
    __syncthreads();
    if (nt + 1 < NTILES_PER_CHUNK)
      issue_tile_async(cbb, n0base + (nt + 1) * 16,
                       ldsBase + (((unsigned)(nt + 1) & 1u) << 13), tid);

    const unsigned short* bbuf = &ldsB[nt & 1][0];
    v8f acc = {0.f, 0.f, 0.f, 0.f, 0.f, 0.f, 0.f, 0.f};
    #pragma unroll
    for (int ks = 0; ks < 8; ++ks) {
      FragBF16 b;
      const unsigned short* p0 = bbuf + m * DC + ks * 32 + half * 8;
      b.q[0] = *reinterpret_cast<const u32x4*>(p0);      // ds_load_b128
      b.q[1] = *reinterpret_cast<const u32x4*>(p0 + 16);
      acc = __builtin_amdgcn_wmma_f32_16x16x32_bf16(
          false, a[ks].v, false, b.v, (short)0, acc, false, false);
    }
    const int col = n0base + nt * 16 + m;   // ascending -> strict '>' keeps lowest idx
    #pragma unroll
    for (int r = 0; r < 8; ++r) {
      float v = acc[r];
      if (v > bestv[r]) { bestv[r] = v; besti[r] = col; }
    }
  }

  // reduce across the 16 lanes of each half (masks <= 8 never cross halves)
  #pragma unroll
  for (int r = 0; r < 8; ++r) {
    float v = bestv[r]; int i = besti[r];
    #pragma unroll
    for (int mask = 8; mask >= 1; mask >>= 1) {
      float ov = __shfl_xor(v, mask, 32);
      int   oi = __shfl_xor(i, mask, 32);
      if (ov > v || (ov == v && oi < i)) { v = ov; i = oi; }
    }
    if (m == 0) {
      int row = mtile * 16 + r + half * 8;
      if (row < P) {
        pval[(size_t)row * NCHUNK + chunk] = v;
        pidx[(size_t)row * NCHUNK + chunk] = i;
      }
    }
  }
}

// 6) reduce chunks -> final index per row; also emit float index to output
__global__ void reduce_idx_kernel(const float* __restrict__ pval, const int* __restrict__ pidx,
                                  int* __restrict__ idx, float* __restrict__ out_idx, int P) {
  int p = blockIdx.x * blockDim.x + threadIdx.x;
  if (p >= P) return;
  float bv = pval[(size_t)p * NCHUNK]; int bi = pidx[(size_t)p * NCHUNK];
  for (int c = 1; c < NCHUNK; ++c) {
    float v = pval[(size_t)p * NCHUNK + c]; int i = pidx[(size_t)p * NCHUNK + c];
    if (v > bv || (v == bv && i < bi)) { bv = v; bi = i; }
  }
  idx[p] = bi;
  out_idx[p] = (float)bi;
}

// 7) commit-loss partial sums (fixed 64 blocks, deterministic order)
__global__ void commit_kernel(const float* __restrict__ rk, const float* __restrict__ cbn,
                              const int* __restrict__ idx, float* __restrict__ partial,
                              int S) {
  int N = DC * S;
  float s = 0.f;
  for (int t = blockIdx.x * blockDim.x + threadIdx.x; t < N; t += gridDim.x * blockDim.x) {
    int d = t / S, p = t % S;
    float diff = rk[t] - cbn[(size_t)idx[p] * DC + d];
    s += diff * diff;
  }
  __shared__ float red[DC];
  red[threadIdx.x] = s; __syncthreads();
  for (int t = 128; t > 0; t >>= 1) { if (threadIdx.x < t) red[threadIdx.x] += red[threadIdx.x + t]; __syncthreads(); }
  if (threadIdx.x == 0) partial[blockIdx.x] = red[0];
}

// 8) nearest-upsample quantized map; accumulate quantized_sum; new residual
__global__ void quantize_kernel(const float* __restrict__ cbn, const int* __restrict__ idx,
                                const float* __restrict__ x, float* __restrict__ qs,
                                float* __restrict__ r, int sh, int sw, int first) {
  int t = blockIdx.x * blockDim.x + threadIdx.x;
  if (t >= DC * HWC * HWC) return;
  int d = t >> 12;
  int yx = t & 4095;
  int y = yx >> 6, xx = yx & 63;
  int f = HWC / sh;
  int code = idx[(y / f) * sw + (xx / f)];
  float q = cbn[(size_t)code * DC + d];
  float s = (first ? 0.f : qs[t]) + q;
  qs[t] = s;
  r[t]  = x[t] - s;
}

// ---------------- deterministic segment sums ----------------
__global__ void zero_count_kernel(int* __restrict__ count) {
  int n = blockIdx.x * blockDim.x + threadIdx.x;
  if (n < KC) count[n] = 0;
}
__global__ void count_kernel(const int* __restrict__ idx, int* __restrict__ count, int P) {
  int p = blockIdx.x * blockDim.x + threadIdx.x;
  if (p < P) atomicAdd(&count[idx[p]], 1);     // integer atomics: deterministic
}
__global__ void scan_kernel(const int* __restrict__ count, int* __restrict__ offset) {
  __shared__ int sums[DC];
  __shared__ int pre[DC + 1];
  int tid = threadIdx.x;
  int base = tid * 32;
  int local = 0;
  for (int i = 0; i < 32; ++i) local += count[base + i];
  sums[tid] = local; __syncthreads();
  if (tid == 0) { pre[0] = 0; for (int i = 0; i < DC; ++i) pre[i + 1] = pre[i] + sums[i]; }
  __syncthreads();
  int run = pre[tid];
  for (int i = 0; i < 32; ++i) { offset[base + i] = run; run += count[base + i]; }
}
__global__ void build_list_kernel(const int* __restrict__ idx, const int* __restrict__ offset,
                                  int* __restrict__ list, int P) {
  int p = blockIdx.x * blockDim.x + threadIdx.x;
  if (p >= P) return;
  int code = idx[p];
  int rank = 0;
  for (int q = 0; q < p; ++q) rank += (idx[q] == code);   // stable rank
  list[offset[code] + rank] = p;
}
__global__ void code_sum_kernel(const float* __restrict__ flat, const int* __restrict__ count,
                                const int* __restrict__ offset, const int* __restrict__ list,
                                float* __restrict__ cb_sum) {
  int n = blockIdx.x, d = threadIdx.x;
  int c = count[n], off = offset[n];
  float s = 0.f;
  for (int j = 0; j < c; ++j) s += flat[(size_t)list[off + j] * DC + d];
  cb_sum[(size_t)n * DC + d] = s;
}

// ---------------- EMA updates ----------------
__global__ void ema_cs_kernel(const float* __restrict__ cs_in, const int* __restrict__ count,
                              float* __restrict__ out_cs, float* __restrict__ partials) {
  float local = 0.f;
  for (int n = blockIdx.x * blockDim.x + threadIdx.x; n < KC; n += gridDim.x * blockDim.x) {
    float v = 0.99f * cs_in[n] + 0.01f * (float)count[n];
    out_cs[n] = v;
    local += v;
  }
  __shared__ float red[DC];
  red[threadIdx.x] = local; __syncthreads();
  for (int t = 128; t > 0; t >>= 1) { if (threadIdx.x < t) red[threadIdx.x] += red[threadIdx.x + t]; __syncthreads(); }
  if (threadIdx.x == 0) partials[blockIdx.x] = red[0];
}
__global__ void ntot_kernel(const float* __restrict__ partials, float* __restrict__ ntot, int nb) {
  if (threadIdx.x == 0 && blockIdx.x == 0) {
    float s = 0.f;
    for (int i = 0; i < nb; ++i) s += partials[i];
    *ntot = s;
  }
}
__global__ void ema_avg_kernel(const float* __restrict__ avg_in, const float* __restrict__ cb_sum,
                               const float* __restrict__ out_cs, const float* __restrict__ ntot_p,
                               float* __restrict__ out_avg, float* __restrict__ out_cb) {
  int t = blockIdx.x * blockDim.x + threadIdx.x;
  if (t >= KC * DC) return;
  int n = t >> 8;
  float navg = 0.99f * avg_in[t] + 0.01f * cb_sum[t];
  out_avg[t] = navg;
  float ntot = *ntot_p;
  float smooth = (out_cs[n] + 1e-5f) / (ntot + (float)KC * 1e-5f) * ntot;
  out_cb[t] = navg / smooth;
}

// final commit loss (fixed order, deterministic)
__global__ void loss_kernel(const float* __restrict__ commit_part, float* __restrict__ out_loss) {
  if (threadIdx.x == 0 && blockIdx.x == 0) {
    const int shs[7] = {1, 2, 4, 8, 16, 32, 64};
    float total = 0.f;
    for (int k = 0; k < 7; ++k) {
      float s = 0.f;
      for (int b = 0; b < 64; ++b) s += commit_part[k * 64 + b];
      total += s / ((float)DC * (float)shs[k] * (float)shs[k]);
    }
    *out_loss = total;
  }
}

// ---------------------------------------------------------------------------
// host side
// ---------------------------------------------------------------------------
extern "C" void kernel_launch(void* const* d_in, const int* in_sizes, int n_in,
                              void* d_out, int out_size, void* d_ws, size_t ws_size,
                              hipStream_t stream) {
  (void)in_sizes; (void)n_in; (void)out_size; (void)ws_size;

  const float* x         = (const float*)d_in[0];  // [256,64,64]
  const float* codebooks = (const float*)d_in[1];  // [7,8192,256]
  const float* cb_avgs   = (const float*)d_in[2];  // [7,8192,256]
  const float* clus_sz   = (const float*)d_in[3];  // [7,8192]
  const float* conv_w    = (const float*)d_in[4];  // [7,256,256,3,3]
  const float* conv_b    = (const float*)d_in[5];  // [7,256]
  float* out = (float*)d_out;
  float* ws  = (float*)d_ws;

  // workspace layout (float units)
  constexpr size_t W_R      = 0;                          // 256*64*64
  constexpr size_t W_RK     = W_R      + 1048576;         // resized residual
  constexpr size_t W_RKC    = W_RK     + 1048576;         // conv output
  constexpr size_t W_FLAT   = W_RKC    + 1048576;         // fp32 normalized rows [4096,256]
  constexpr size_t W_FLATB  = W_FLAT   + 1048576;         // bf16 rows (524288 f-units)
  constexpr size_t W_CBN    = W_FLATB  + 524288;          // fp32 normalized codebook [8192,256]
  constexpr size_t W_CBNB   = W_CBN    + 2097152;         // bf16 codebook (1048576 f-units)
  constexpr size_t W_PVAL   = W_CBNB   + 1048576;         // [4096,16] partial max
  constexpr size_t W_PIDX   = W_PVAL   + 65536;           // [4096,16] partial idx (int)
  constexpr size_t W_IDX    = W_PIDX   + 65536;           // [4096] (int)
  constexpr size_t W_CBSUM  = W_IDX    + 4096;            // [8192,256]
  constexpr size_t W_COUNT  = W_CBSUM  + 2097152;         // [8192] (int)
  constexpr size_t W_OFFSET = W_COUNT  + 8192;            // [8192] (int)
  constexpr size_t W_LIST   = W_OFFSET + 8192;            // [4096] (int)
  constexpr size_t W_COMMIT = W_LIST   + 4096;            // [7*64]
  constexpr size_t W_PART   = W_COMMIT + 448;             // [32]
  constexpr size_t W_NTOT   = W_PART   + 32;              // [1]

  // output layout (float units)
  constexpr size_t QS_SIZE  = (size_t)DC * HWC * HWC;           // 1,048,576
  constexpr size_t UPD_SIZE = (size_t)KC + 2 * (size_t)KC * DC; // 4,202,496
  constexpr size_t IDX_BASE = QS_SIZE + 7 * UPD_SIZE;
  const int shs[7] = {1, 2, 4, 8, 16, 32, 64};
  const size_t idx_pref[7] = {0, 1, 5, 21, 85, 341, 1365};
  constexpr size_t LOSS_OFF = IDX_BASE + 5461;

  for (int k = 0; k < 7; ++k) {
    const int sh = shs[k], sw = shs[k];
    const int S = sh * sw;
    const int P = S;
    const int Mtiles = (P + 15) / 16;
    const int Mpad = Mtiles * 16;

    // 1) resize residual (k==0 reads x directly)
    {
      int N = DC * S;
      resize_kernel<<<(N + 255) / 256, 256, 0, stream>>>(
          (k == 0) ? x : (ws + W_R), ws + W_RK, sh, sw);
    }
    // 2) conv3x3
    conv3x3_kernel<<<S, DC, 0, stream>>>(
        ws + W_RK, conv_w + (size_t)k * DC * DC * 9, conv_b + (size_t)k * DC,
        ws + W_RKC, sh, sw);
    // 3) flatten + normalize rows (pad to Mpad)
    norm_flat_kernel<<<Mpad, DC, 0, stream>>>(
        ws + W_RKC, ws + W_FLAT, (unsigned short*)(ws + W_FLATB), P, S);
    // 4) normalize codebook k
    cbnorm_kernel<<<KC, DC, 0, stream>>>(
        codebooks + (size_t)k * KC * DC, ws + W_CBN, (unsigned short*)(ws + W_CBNB));
    // 5) WMMA scoring + chunk argmax (4 waves share LDS-staged B tiles)
    {
      dim3 grid(NCHUNK, (Mtiles + 3) / 4);
      score_argmax_kernel<<<grid, 128, 0, stream>>>(
          (const unsigned short*)(ws + W_FLATB), (const unsigned short*)(ws + W_CBNB),
          ws + W_PVAL, (int*)(ws + W_PIDX), P);
    }
    // 6) final argmax + index output
    reduce_idx_kernel<<<(P + 255) / 256, 256, 0, stream>>>(
        ws + W_PVAL, (const int*)(ws + W_PIDX), (int*)(ws + W_IDX),
        out + IDX_BASE + idx_pref[k], P);
    // 7) commit-loss partials
    commit_kernel<<<64, DC, 0, stream>>>(
        ws + W_RKC, ws + W_CBN, (const int*)(ws + W_IDX), ws + W_COMMIT + (size_t)k * 64, S);
    // 8) quantized_sum += upsample(z_q); r = x - quantized_sum
    quantize_kernel<<<(DC * HWC * HWC + 255) / 256, 256, 0, stream>>>(
        ws + W_CBN, (const int*)(ws + W_IDX), x, out /*quantized_sum*/, ws + W_R,
        sh, sw, (k == 0) ? 1 : 0);
    // 9) deterministic segment sums for EMA
    zero_count_kernel<<<KC / 256, 256, 0, stream>>>((int*)(ws + W_COUNT));
    count_kernel<<<(P + 255) / 256, 256, 0, stream>>>(
        (const int*)(ws + W_IDX), (int*)(ws + W_COUNT), P);
    scan_kernel<<<1, DC, 0, stream>>>((const int*)(ws + W_COUNT), (int*)(ws + W_OFFSET));
    build_list_kernel<<<(P + 255) / 256, 256, 0, stream>>>(
        (const int*)(ws + W_IDX), (const int*)(ws + W_OFFSET), (int*)(ws + W_LIST), P);
    code_sum_kernel<<<KC, DC, 0, stream>>>(
        ws + W_FLAT, (const int*)(ws + W_COUNT), (const int*)(ws + W_OFFSET),
        (const int*)(ws + W_LIST), ws + W_CBSUM);
    // 10) EMA: new_cs, n_tot, new_avg, new_cb
    float* out_cs  = out + QS_SIZE + (size_t)k * UPD_SIZE;
    float* out_avg = out_cs + KC;
    float* out_cb  = out_avg + (size_t)KC * DC;
    ema_cs_kernel<<<32, DC, 0, stream>>>(
        clus_sz + (size_t)k * KC, (const int*)(ws + W_COUNT), out_cs, ws + W_PART);
    ntot_kernel<<<1, 32, 0, stream>>>(ws + W_PART, ws + W_NTOT, 32);
    ema_avg_kernel<<<(KC * DC) / 256, 256, 0, stream>>>(
        cb_avgs + (size_t)k * KC * DC, ws + W_CBSUM, out_cs, ws + W_NTOT, out_avg, out_cb);
  }

  // commit loss
  loss_kernel<<<1, 32, 0, stream>>>(ws + W_COMMIT, out + LOSS_OFF);
}
